// MambaLM_80444737454250
// MI455X (gfx1250) — matrix operand order repaired and checked by reading, not compiled, not run on hardware
//
#include <hip/hip_runtime.h>
#include <hip/hip_bf16.h>
#include <math.h>

// ---------------------------------------------------------------------------
// Model dimensions (fixed by the reference)
// ---------------------------------------------------------------------------
#define S_    1024
#define D_    768
#define NL_   2
#define N16   16
#define K4    4
#define DI_   1536           // E*D
#define DTR   48             // D/16
#define XDL   80             // DT_RANK + 2*N
#define V_    32000

typedef __attribute__((ext_vector_type(16))) _Float16 v16h;
typedef __attribute__((ext_vector_type(8)))  float    v8f;
typedef __attribute__((ext_vector_type(4)))  float    fx4;

// ---------------------------------------------------------------------------
// Fragment loaders: fp32 memory -> f16 WMMA fragments (contiguous k order k = 16*(lane/16) + s, same in A and B)
// A [M,K] row-major:  lane(0-15)=row m, K=[k0..k0+15]
//                     lane(16-31)=row m, K=[k0+16..k0+31]
// B as W [N,K] row-major (WMMA B = K x N = W^T):
//                     lane(0-15)=col n, K=[k0..k0+15]; lane(16-31)=col n, K=[k0+16..k0+31]
// Requires N % 16 == 0 for all GEMMs in this model (true: 3072/80/1536/768/32000).
// ---------------------------------------------------------------------------
__device__ __forceinline__ v16h load_frag_a_full(const float* __restrict__ A, int lda,
                                                 int arow, int k0, int lane)
{
    v16h r;
    const int off0 = (lane < 16) ? 0 : 16;
    const float* p = A + (size_t)arow * lda + k0 + off0;
    fx4 c0 = *(const fx4*)(p);
    fx4 c1 = *(const fx4*)(p + 4);
    fx4 c2 = *(const fx4*)(p + 8);
    fx4 c3 = *(const fx4*)(p + 12);
#pragma unroll
    for (int i = 0; i < 4; ++i) {
        r[i]      = (_Float16)c0[i];
        r[4 + i]  = (_Float16)c1[i];
        r[8 + i]  = (_Float16)c2[i];
        r[12 + i] = (_Float16)c3[i];
    }
    return r;
}

__device__ __forceinline__ v16h load_frag_a_tail(const float* __restrict__ A, int lda,
                                                 int arow, int k0, int lane, int K)
{
    v16h r;
    const int off0 = (lane < 16) ? 0 : 16;
#pragma unroll
    for (int i = 0; i < 16; ++i) {
        int kk = k0 + off0 + i;
        float v = (kk < K) ? A[(size_t)arow * lda + kk] : 0.0f;
        r[i] = (_Float16)v;
    }
    return r;
}

__device__ __forceinline__ v16h load_frag_b_full(const float* __restrict__ B, int ldb,
                                                 int brow, int k0, int lane)
{
    v16h r;
    const int kb = k0 + ((lane < 16) ? 0 : 16);
    const float* p = B + (size_t)brow * ldb + kb;
    __builtin_prefetch(p + 32, 0, 1);        // next K-chunk of this weight row
    fx4 c0 = *(const fx4*)(p);
    fx4 c1 = *(const fx4*)(p + 4);
    fx4 c2 = *(const fx4*)(p + 8);
    fx4 c3 = *(const fx4*)(p + 12);
#pragma unroll
    for (int i = 0; i < 4; ++i) {
        r[i]      = (_Float16)c0[i];
        r[4 + i]  = (_Float16)c1[i];
        r[8 + i]  = (_Float16)c2[i];
        r[12 + i] = (_Float16)c3[i];
    }
    return r;
}

__device__ __forceinline__ v16h load_frag_b_tail(const float* __restrict__ B, int ldb,
                                                 int brow, int k0, int lane, int K)
{
    v16h r;
    const int kb = k0 + ((lane < 16) ? 0 : 16);
#pragma unroll
    for (int i = 0; i < 16; ++i) {
        int kk = kb + i;
        float v = (kk < K) ? B[(size_t)brow * ldb + kk] : 0.0f;
        r[i] = (_Float16)v;
    }
    return r;
}

__device__ __forceinline__ v8f wmma_f16(v16h a, v16h b, v8f c)
{
    return __builtin_amdgcn_wmma_f32_16x16x32_f16(false, a, false, b,
                                                  (short)0, c, false, false);
}

// ---------------------------------------------------------------------------
// C[M,N] = A[M,K] * W[N,K]^T   (fp32 memory, f16 WMMA math, f32 accumulate)
// block = 256 thr = 8 waves; wave tile = 16(M) x 64(N); block tile = 32 x 256.
// All tile-selection control flow is SGPR-based (readfirstlane) so EXEC is
// all-ones at every WMMA (ISA 7.12 requirement).
// EVEN_N path: all 4 B fragments loaded into distinct live ranges BEFORE the
// WMMAs -> one big s_clause of b128 loads (staged waitcnts, full MLP), then
// 4 back-to-back v_wmma; unroll-by-2 lets next iter's loads hide under them.
// ---------------------------------------------------------------------------
template <bool EVEN_N, bool EVEN_K>
__global__ __launch_bounds__(256)
void k_gemm_wmma(const float* __restrict__ A, int lda,
                 const float* __restrict__ B, int ldb,
                 float* __restrict__ C, int ldc,
                 int M, int N, int K)
{
    const int lane   = threadIdx.x & 31;
    const int wave   = __builtin_amdgcn_readfirstlane(threadIdx.x >> 5);  // SGPR
    const int tile_m = blockIdx.y * 32 + (wave >> 2) * 16;                // SGPR
    const int tile_n = blockIdx.x * 256 + (wave & 3) * 64;                // SGPR
    if (tile_m >= M) return;                                              // scalar

    const v8f vzero = {0.f, 0.f, 0.f, 0.f, 0.f, 0.f, 0.f, 0.f};
    v8f acc0 = vzero, acc1 = vzero, acc2 = vzero, acc3 = vzero;

    const int arow  = tile_m + (lane & 15);
    const int ncol  = lane & 15;
    const int kfull = EVEN_K ? K : (K & ~31);

    if (EVEN_N) {
#pragma unroll 2
        for (int k0 = 0; k0 < kfull; k0 += 32) {
            // issue ALL loads first: 4 (A) + 16 (B) b128 loads cluster into
            // s_clauses; f32->f16 cvts overlap the outstanding loads.
            v16h a  = load_frag_a_full(A, lda, arow, k0, lane);
            v16h b0 = load_frag_b_full(B, ldb, tile_n +  0 + ncol, k0, lane);
            v16h b1 = load_frag_b_full(B, ldb, tile_n + 16 + ncol, k0, lane);
            v16h b2 = load_frag_b_full(B, ldb, tile_n + 32 + ncol, k0, lane);
            v16h b3 = load_frag_b_full(B, ldb, tile_n + 48 + ncol, k0, lane);
            // back-to-back WMMAs: distinct D regs, shared sources -> no hazard
            acc0 = wmma_f16(a, b0, acc0);
            acc1 = wmma_f16(a, b1, acc1);
            acc2 = wmma_f16(a, b2, acc2);
            acc3 = wmma_f16(a, b3, acc3);
        }
        if (!EVEN_K && kfull < K) {
            v16h a  = load_frag_a_tail(A, lda, arow, kfull, lane, K);
            v16h b0 = load_frag_b_tail(B, ldb, tile_n +  0 + ncol, kfull, lane, K);
            v16h b1 = load_frag_b_tail(B, ldb, tile_n + 16 + ncol, kfull, lane, K);
            v16h b2 = load_frag_b_tail(B, ldb, tile_n + 32 + ncol, kfull, lane, K);
            v16h b3 = load_frag_b_tail(B, ldb, tile_n + 48 + ncol, kfull, lane, K);
            acc0 = wmma_f16(a, b0, acc0);
            acc1 = wmma_f16(a, b1, acc1);
            acc2 = wmma_f16(a, b2, acc2);
            acc3 = wmma_f16(a, b3, acc3);
        }
    } else {
        // ragged-N path (x_proj GEMM only, N=80): scalar per-sub-tile guards
        v8f* accp[4] = { &acc0, &acc1, &acc2, &acc3 };
        for (int k0 = 0; k0 < kfull; k0 += 32) {
            v16h a = load_frag_a_full(A, lda, arow, k0, lane);
#pragma unroll
            for (int j = 0; j < 4; ++j) {
                if ((tile_n + j * 16) < N) {                              // scalar
                    v16h b = load_frag_b_full(B, ldb, tile_n + j * 16 + ncol, k0, lane);
                    *accp[j] = wmma_f16(a, b, *accp[j]);
                }
            }
        }
        if (!EVEN_K && kfull < K) {
            v16h a = load_frag_a_tail(A, lda, arow, kfull, lane, K);
#pragma unroll
            for (int j = 0; j < 4; ++j) {
                if ((tile_n + j * 16) < N) {                              // scalar
                    v16h b = load_frag_b_tail(B, ldb, tile_n + j * 16 + ncol, kfull, lane, K);
                    *accp[j] = wmma_f16(a, b, *accp[j]);
                }
            }
        }
    }

    // D layout: lane(0-15): N=lane, M=tile_m+r ; lane(16-31): N=lane-16, M=tile_m+8+r
    const int rbase = tile_m + ((lane < 16) ? 0 : 8);
    const v8f accs[4] = { acc0, acc1, acc2, acc3 };
#pragma unroll
    for (int j = 0; j < 4; ++j) {
        if (EVEN_N || (tile_n + j * 16) < N) {                            // scalar
            float* cp = C + (size_t)rbase * ldc + tile_n + j * 16 + ncol;
#pragma unroll
            for (int r = 0; r < 8; ++r)
                cp[(size_t)r * ldc] = accs[j][r];
        }
    }
}

// ---------------------------------------------------------------------------
// Embedding gather
// ---------------------------------------------------------------------------
__global__ __launch_bounds__(256)
void k_embed(const int* __restrict__ tok, const float* __restrict__ emb,
             float* __restrict__ x)
{
    int idx = blockIdx.x * 256 + threadIdx.x;
    if (idx >= S_ * D_) return;
    int t = idx / D_, d = idx - t * D_;
    x[idx] = emb[(size_t)tok[t] * D_ + d];
}

// ---------------------------------------------------------------------------
// Depthwise causal conv1d (K=4) + bias + SiLU.  xs = xz[:, :DI]
// ---------------------------------------------------------------------------
__global__ __launch_bounds__(256)
void k_conv_silu(const float* __restrict__ xz, const float* __restrict__ cw,
                 const float* __restrict__ cb, float* __restrict__ xs)
{
    int idx = blockIdx.x * 256 + threadIdx.x;
    if (idx >= S_ * DI_) return;
    int t = idx / DI_, c = idx - t * DI_;
    float s = cb[c];
#pragma unroll
    for (int j = 0; j < K4; ++j) {
        int tt = t + j - (K4 - 1);
        if (tt >= 0) s += cw[c * K4 + j] * xz[(size_t)tt * (2 * DI_) + c];
    }
    float sig = 1.0f / (1.0f + __expf(-s));
    xs[idx] = s * sig;
}

// ---------------------------------------------------------------------------
// Selective scan: thread = channel; h[16] in VGPRs; B/C broadcast via LDS
// (double-buffered, one barrier per timestep).  Fuses softplus(dt) + D-skip
// + SiLU(z) gate.
// ---------------------------------------------------------------------------
__global__ __launch_bounds__(256)
void k_scan(const float* __restrict__ dtp, const float* __restrict__ dtb,
            const float* __restrict__ xz,  const float* __restrict__ xs,
            const float* __restrict__ xdbl,const float* __restrict__ alog,
            const float* __restrict__ dpar, float* __restrict__ y)
{
    __shared__ float sBC[2][2 * N16];
    const int c = blockIdx.x * 256 + threadIdx.x;       // 0..DI-1 (6 blocks x 256)
    float a[N16], h[N16];
#pragma unroll
    for (int n = 0; n < N16; ++n) { a[n] = -__expf(alog[c * N16 + n]); h[n] = 0.f; }
    const float db = dtb[c], dskip = dpar[c];

    for (int t = 0; t < S_; ++t) {
        const int buf = t & 1;
        if (threadIdx.x < 2 * N16)
            sBC[buf][threadIdx.x] = xdbl[t * XDL + DTR + threadIdx.x];
        __syncthreads();

        float d  = dtp[t * DI_ + c] + db;
        float dt = (d > 20.f) ? d : log1pf(__expf(d));   // softplus
        float xv = xs[t * DI_ + c];
        float zv = xz[(size_t)t * (2 * DI_) + DI_ + c];
        float dtx = dt * xv;
        float acc = 0.f;
#pragma unroll
        for (int n = 0; n < N16; ++n) {
            float dA = __expf(dt * a[n]);                // v_exp_f32 (TRANS)
            h[n] = dA * h[n] + dtx * sBC[buf][n];        // B
            acc += h[n] * sBC[buf][N16 + n];             // C
        }
        acc += xv * dskip;
        float sig = 1.0f / (1.0f + __expf(-zv));
        y[t * DI_ + c] = acc * (zv * sig);               // * silu(z)
    }
}

// ---------------------------------------------------------------------------
// Residual add + LayerNorm over D=768 (block per row, 3 elems/thread)
// ---------------------------------------------------------------------------
__global__ __launch_bounds__(256)
void k_add_ln(float* __restrict__ x, const float* __restrict__ mo,
              const float* __restrict__ g, const float* __restrict__ b)
{
    __shared__ float sred[256];
    const int row = blockIdx.x;
    const int tid = threadIdx.x;
    float v[3];
    float s = 0.f;
#pragma unroll
    for (int i = 0; i < 3; ++i) {
        int d = tid + i * 256;
        v[i] = x[row * D_ + d] + mo[row * D_ + d];
        s += v[i];
    }
    sred[tid] = s; __syncthreads();
    for (int off = 128; off > 0; off >>= 1) {
        if (tid < off) sred[tid] += sred[tid + off];
        __syncthreads();
    }
    const float mu = sred[0] * (1.0f / D_);
    __syncthreads();
    float vs = 0.f;
#pragma unroll
    for (int i = 0; i < 3; ++i) { float dl = v[i] - mu; vs += dl * dl; }
    sred[tid] = vs; __syncthreads();
    for (int off = 128; off > 0; off >>= 1) {
        if (tid < off) sred[tid] += sred[tid + off];
        __syncthreads();
    }
    const float rstd = rsqrtf(sred[0] * (1.0f / D_) + 1e-5f);
#pragma unroll
    for (int i = 0; i < 3; ++i) {
        int d = tid + i * 256;
        x[row * D_ + d] = (v[i] - mu) * rstd * g[d] + b[d];
    }
}

// ---------------------------------------------------------------------------
// Launcher
// ---------------------------------------------------------------------------
extern "C" void kernel_launch(void* const* d_in, const int* in_sizes, int n_in,
                              void* d_out, int out_size, void* d_ws, size_t ws_size,
                              hipStream_t stream)
{
    (void)in_sizes; (void)n_in; (void)out_size; (void)ws_size;

    const int*   tokens = (const int*)  d_in[0];
    const float* emb    = (const float*)d_in[1];
    const float* head_w = (const float*)d_in[13];

    // workspace layout (floats)
    float* ws     = (float*)d_ws;
    float* x      = ws;                                  // [S,D]
    float* xz     = x      + (size_t)S_ * D_;            // [S,2*DI]
    float* xs_act = xz     + (size_t)S_ * 2 * DI_;       // [S,DI]
    float* x_dbl  = xs_act + (size_t)S_ * DI_;           // [S,80]
    float* dtproj = x_dbl  + (size_t)S_ * XDL;           // [S,DI]
    float* y_act  = dtproj + (size_t)S_ * DI_;           // [S,DI]
    float* mo     = y_act  + (size_t)S_ * DI_;           // [S,D]

    auto gemm = [&](const float* A, int lda, const float* B, int ldb,
                    float* C, int ldc, int M, int N, int K) {
        dim3 grid((N + 255) / 256, (M + 31) / 32);
        const bool evenN = (N % 256) == 0;
        const bool evenK = (K % 32) == 0;
        if (evenN && evenK)
            k_gemm_wmma<true,  true ><<<grid, 256, 0, stream>>>(A, lda, B, ldb, C, ldc, M, N, K);
        else if (evenN)
            k_gemm_wmma<true,  false><<<grid, 256, 0, stream>>>(A, lda, B, ldb, C, ldc, M, N, K);
        else if (evenK)
            k_gemm_wmma<false, true ><<<grid, 256, 0, stream>>>(A, lda, B, ldb, C, ldc, M, N, K);
        else
            k_gemm_wmma<false, false><<<grid, 256, 0, stream>>>(A, lda, B, ldb, C, ldc, M, N, K);
    };

    k_embed<<<(S_ * D_ + 255) / 256, 256, 0, stream>>>(tokens, emb, x);

    for (int l = 0; l < NL_; ++l) {
        const float* in_w    = (const float*)d_in[2]  + (size_t)l * 2 * DI_ * D_;
        const float* conv_w  = (const float*)d_in[3]  + (size_t)l * DI_ * K4;
        const float* conv_b  = (const float*)d_in[4]  + (size_t)l * DI_;
        const float* xproj_w = (const float*)d_in[5]  + (size_t)l * XDL * DI_;
        const float* dt_w    = (const float*)d_in[6]  + (size_t)l * DI_ * DTR;
        const float* dt_b    = (const float*)d_in[7]  + (size_t)l * DI_;
        const float* A_log   = (const float*)d_in[8]  + (size_t)l * DI_ * N16;
        const float* Dp      = (const float*)d_in[9]  + (size_t)l * DI_;
        const float* out_w   = (const float*)d_in[10] + (size_t)l * D_ * DI_;
        const float* ln_g    = (const float*)d_in[11] + (size_t)l * D_;
        const float* ln_b    = (const float*)d_in[12] + (size_t)l * D_;

        // xz = x @ in_w^T            [1024,3072] K=768   (EVEN_N, EVEN_K)
        gemm(x, D_, in_w, D_, xz, 2 * DI_, S_, 2 * DI_, D_);
        // depthwise conv + SiLU
        k_conv_silu<<<(S_ * DI_ + 255) / 256, 256, 0, stream>>>(xz, conv_w, conv_b, xs_act);
        // x_dbl = xs @ xproj_w^T     [1024,80] K=1536    (ragged N, EVEN_K)
        gemm(xs_act, DI_, xproj_w, DI_, x_dbl, XDL, S_, XDL, DI_);
        // dtproj = x_dbl[:, :48] @ dt_w^T  [1024,1536] K=48  (EVEN_N, ragged K)
        gemm(x_dbl, XDL, dt_w, DTR, dtproj, DI_, S_, DI_, DTR);
        // selective scan (fused softplus/D-skip/SiLU gate)
        k_scan<<<DI_ / 256, 256, 0, stream>>>(dtproj, dt_b, xz, xs_act, x_dbl,
                                              A_log, Dp, y_act);
        // mamba_out = y @ out_w^T    [1024,768] K=1536   (EVEN_N, EVEN_K)
        gemm(y_act, DI_, out_w, DI_, mo, D_, S_, D_, DI_);
        // x = LN(x + mamba_out)
        k_add_ln<<<S_, 256, 0, stream>>>(x, mo, ln_g, ln_b);
    }

    // logits = x @ head_w^T          [1024,32000] K=768  (EVEN_N, EVEN_K, dominant)
    gemm(x, D_, head_w, D_, (float*)d_out, V_, S_, V_, D_);
}
